// Decoder_Block_62130996904466
// MI455X (gfx1250) — matrix operand order repaired and checked by reading, not compiled, not run on hardware
//
#include <hip/hip_runtime.h>

typedef __attribute__((ext_vector_type(16))) _Float16 v16h;
typedef __attribute__((ext_vector_type(8)))  _Float16 v8h;
typedef __attribute__((ext_vector_type(4)))  _Float16 v4h;
typedef __attribute__((ext_vector_type(8)))  float    v8f;
typedef __attribute__((ext_vector_type(4)))  float    v4f;

union Frag16 { v16h v; v8h h[2]; };

__device__ __forceinline__ v8f wmma16x16x32(const v16h a, const v16h b, const v8f c) {
  // D = A(16x32 f16) * B(32x16 f16) + C(16x16 f32)
  return __builtin_amdgcn_wmma_f32_16x16x32_f16(false, a, false, b, (short)0, c, false, false);
}

// ---- CDNA5 async global->LDS copy (ASYNCcnt-tracked, no VGPR staging) ------
__device__ __forceinline__ void async_copy_b128(const void* gptr, void* lptr) {
  unsigned loff = (unsigned)(unsigned long long)lptr;  // low 32 bits = LDS offset
  asm volatile("global_load_async_to_lds_b128 %0, %1, off"
               :: "v"(loff), "v"(gptr) : "memory");
}
__device__ __forceinline__ void wait_async() {
  asm volatile("s_wait_asynccnt 0x0" ::: "memory");
}

__device__ __forceinline__ float half16_max(float v) {
  #pragma unroll
  for (int m = 1; m < 16; m <<= 1) v = fmaxf(v, __shfl_xor(v, m, 32));
  return v;
}
__device__ __forceinline__ float half16_sum(float v) {
  #pragma unroll
  for (int m = 1; m < 16; m <<= 1) v += __shfl_xor(v, m, 32);
  return v;
}

// ---------------------------------------------------------------------------
// f32 -> f16 elementwise convert (vector4)
// ---------------------------------------------------------------------------
__global__ __launch_bounds__(256)
void cvt_f16_kernel(const float* __restrict__ in, _Float16* __restrict__ out, int n4) {
  int i = blockIdx.x * 256 + threadIdx.x;
  if (i >= n4) return;
  v4f x = *(const v4f*)(in + (size_t)i * 4);
  v4h h; h[0] = (_Float16)x[0]; h[1] = (_Float16)x[1]; h[2] = (_Float16)x[2]; h[3] = (_Float16)x[3];
  *(v4h*)(out + (size_t)i * 4) = h;
}

// ---------------------------------------------------------------------------
// W[K,Nw] f32  ->  Wt[(noff+n)*ldo + k] f16   (transpose + downconvert)
// grid: (K/16, Nw/16), block 256 (16x16)
// ---------------------------------------------------------------------------
__global__ __launch_bounds__(256)
void cvt_tr_kernel(const float* __restrict__ W, _Float16* __restrict__ out,
                   int K, int Nw, int noff, int ldo) {
  __shared__ float tile[16][17];
  int k0 = blockIdx.x * 16, n0 = blockIdx.y * 16;
  int tx = threadIdx.x & 15, ty = threadIdx.x >> 4;
  tile[ty][tx] = W[(size_t)(k0 + ty) * Nw + (n0 + tx)];
  __syncthreads();
  out[(size_t)(noff + n0 + ty) * ldo + (k0 + tx)] = (_Float16)tile[tx][ty];
}

// ---------------------------------------------------------------------------
// WMMA GEMM: C[M,N] = A[M,K](f16) * Wt[N,K](f16)^T + bias (+resid) (+relu)
// block tile 128x128, 8 waves of 32x64 (2x4 wmma tiles), K-step 32.
// Double-buffered LDS, tiles filled by global_load_async_to_lds_b128 while
// WMMAs run on the other buffer; one barrier per K-step.
// ---------------------------------------------------------------------------
#define LDA 40
#define LDB 40
__global__ __launch_bounds__(256)
void gemm_wmma_kernel(const _Float16* __restrict__ A, const _Float16* __restrict__ Wt,
                      const float* __restrict__ bias, const float* __restrict__ resid,
                      float* __restrict__ outF, _Float16* __restrict__ outH,
                      int M, int N, int K, int relu) {
  __shared__ __align__(16) _Float16 As[2][128 * LDA];
  __shared__ __align__(16) _Float16 Bs[2][128 * LDB];
  const int tid = threadIdx.x, lane = tid & 31, wave = tid >> 5;
  const int g = lane >> 4, ln = lane & 15;
  const int wm = wave & 3, wn = wave >> 2;           // 32*wm rows, 64*wn cols
  const int mblk = blockIdx.y * 128, nblk = blockIdx.x * 128;

  // tile loader: each thread owns one 16-half slice of a row in A and in B
  const int lrow = tid >> 1, lcol = (tid & 1) * 16;
  const _Float16* agBase = A  + (size_t)(mblk + lrow) * K + lcol;
  const _Float16* bgBase = Wt + (size_t)(nblk + lrow) * K + lcol;

  v8f acc[2][4] = {};

  auto issue = [&](int k0, int buf) {
    async_copy_b128(agBase + k0,     &As[buf][lrow * LDA + lcol]);
    async_copy_b128(agBase + k0 + 8, &As[buf][lrow * LDA + lcol + 8]);
    async_copy_b128(bgBase + k0,     &Bs[buf][lrow * LDB + lcol]);
    async_copy_b128(bgBase + k0 + 8, &Bs[buf][lrow * LDB + lcol + 8]);
  };

  issue(0, 0);
  wait_async();
  __syncthreads();

  int cur = 0;
  for (int k0 = 0; k0 < K; k0 += 32) {
    if (k0 + 32 < K) issue(k0 + 32, cur ^ 1);   // overlap next fill with compute

    Frag16 af[2], bf[4];
    #pragma unroll
    for (int mi = 0; mi < 2; ++mi) {
      int row = wm * 32 + mi * 16 + ln;
      af[mi].h[0] = *(const v8h*)&As[cur][row * LDA + 8 * g];
      af[mi].h[1] = *(const v8h*)&As[cur][row * LDA + 16 + 8 * g];
    }
    #pragma unroll
    for (int ni = 0; ni < 4; ++ni) {
      int n = wn * 64 + ni * 16 + ln;
      bf[ni].h[0] = *(const v8h*)&Bs[cur][n * LDB + 8 * g];
      bf[ni].h[1] = *(const v8h*)&Bs[cur][n * LDB + 16 + 8 * g];
    }
    #pragma unroll
    for (int mi = 0; mi < 2; ++mi)
      #pragma unroll
      for (int ni = 0; ni < 4; ++ni)
        acc[mi][ni] = wmma16x16x32(af[mi].v, bf[ni].v, acc[mi][ni]);

    wait_async();
    __syncthreads();
    cur ^= 1;
  }

  #pragma unroll
  for (int mi = 0; mi < 2; ++mi) {
    #pragma unroll
    for (int ni = 0; ni < 4; ++ni) {
      int ncol = nblk + wn * 64 + ni * 16 + ln;
      float bv = bias ? bias[ncol] : 0.f;
      #pragma unroll
      for (int r = 0; r < 8; ++r) {
        int mrow = mblk + wm * 32 + mi * 16 + r + 8 * g;
        float v = acc[mi][ni][r] + bv;
        if (relu) v = fmaxf(v, 0.f);
        if (resid) v += resid[(size_t)mrow * N + ncol];
        if (outF) outF[(size_t)mrow * N + ncol] = v;
        if (outH) outH[(size_t)mrow * N + ncol] = (_Float16)v;
      }
    }
  }
}

// ---------------------------------------------------------------------------
// Flash attention: QKV packed [B,S,1536] f16 (Q|K|V, 8 heads x 64), causal +
// padding mask, out f32 [B,S,512]. grid (S/128, heads, B), 8 waves x 16 rows.
// K tiles arrive via async global->LDS; scores/PV through f16 WMMA.
// ---------------------------------------------------------------------------
#define LDQ 72
#define LDK 72
#define LDV 40
#define LDP 40
__global__ __launch_bounds__(256)
void flash_attn_kernel(const _Float16* __restrict__ QKV, const float* __restrict__ amask,
                       float* __restrict__ out, int S) {
  __shared__ __align__(16) _Float16 Qs[128 * LDQ];
  __shared__ __align__(16) _Float16 Ks[32 * LDK];
  __shared__ __align__(16) _Float16 Vs[64 * LDV];      // transposed [d][k]
  __shared__ __align__(16) _Float16 Ps[8 * 16 * LDP];  // per-wave P scratch
  __shared__ float Msk[32];

  const int tid = threadIdx.x, lane = tid & 31, wave = tid >> 5;
  const int g = lane >> 4, ln = lane & 15;
  const int bq = blockIdx.x, h = blockIdx.y, b = blockIdx.z;
  const _Float16* Qb = QKV + (size_t)b * S * 1536 + h * 64;
  const _Float16* Kb = Qb + 512;
  const _Float16* Vb = Qb + 1024;

  { // stage Q tile 128x64
    int row = tid >> 1, col = (tid & 1) * 32;
    const _Float16* qg = Qb + (size_t)(bq * 128 + row) * 1536 + col;
    #pragma unroll
    for (int j = 0; j < 4; ++j)
      *(v8h*)&Qs[row * LDQ + col + j * 8] = *(const v8h*)(qg + j * 8);
  }
  __syncthreads();

  Frag16 qf[2];  // Q A-fragments for d chunks 0..31 and 32..63
  {
    int row = wave * 16 + ln;
    #pragma unroll
    for (int c = 0; c < 2; ++c) {
      qf[c].h[0] = *(const v8h*)&Qs[row * LDQ + c * 32 + 8 * g];
      qf[c].h[1] = *(const v8h*)&Qs[row * LDQ + c * 32 + 16 + 8 * g];
    }
  }

  const int q0 = bq * 128 + wave * 16;
  float m[8], l[8];
  v8f accO[4] = {};
  #pragma unroll
  for (int r = 0; r < 8; ++r) { m[r] = -1e30f; l[r] = 0.f; }

  const int nkt = 4 * (bq + 1);
  for (int kt = 0; kt < nkt; ++kt) {
    __syncthreads();
    const int kb = kt * 32;
    { // async co-load K tile [32][64] straight into LDS
      int row = tid >> 3, col = (tid & 7) * 8;
      async_copy_b128(Kb + (size_t)(kb + row) * 1536 + col, &Ks[row * LDK + col]);
    }
    { // co-load V tile transposed -> Vs[d][k]
      int row = tid >> 3, col = (tid & 7) * 8;
      v8h vv = *(const v8h*)(Vb + (size_t)(kb + row) * 1536 + col);
      #pragma unroll
      for (int j = 0; j < 8; ++j) Vs[(col + j) * LDV + row] = vv[j];
    }
    if (tid < 32) Msk[tid] = amask[(size_t)b * S + kb + tid];
    wait_async();
    __syncthreads();

    if (kb > q0 + 15) continue;  // fully above causal diagonal for this wave

    // scores: two 16x16 tiles along 32 keys, K-dim = d (64) in two wmma each
    v8f sc[2];
    #pragma unroll
    for (int sub = 0; sub < 2; ++sub) {
      Frag16 kf0, kf1;
      int kr = sub * 16 + ln;
      kf0.h[0] = *(const v8h*)&Ks[kr * LDK + 8 * g];
      kf0.h[1] = *(const v8h*)&Ks[kr * LDK + 16 + 8 * g];
      kf1.h[0] = *(const v8h*)&Ks[kr * LDK + 32 + 8 * g];
      kf1.h[1] = *(const v8h*)&Ks[kr * LDK + 48 + 8 * g];
      v8f s = {};
      s = wmma16x16x32(qf[0].v, kf0.v, s);
      s = wmma16x16x32(qf[1].v, kf1.v, s);
      sc[sub] = s;
    }

    const int key0 = kb + ln, key1 = kb + 16 + ln;
    const float am0 = Msk[ln], am1 = Msk[16 + ln];
    float rowmax[8];
    #pragma unroll
    for (int r = 0; r < 8; ++r) {
      int q = q0 + r + 8 * g;
      float x0 = sc[0][r] * 0.125f;                 // 1/sqrt(64)
      float x1 = sc[1][r] * 0.125f;
      if (am0 + ((key0 > q) ? 1.f : 0.f) >= 1.f) x0 = -1e30f;
      if (am1 + ((key1 > q) ? 1.f : 0.f) >= 1.f) x1 = -1e30f;
      sc[0][r] = x0; sc[1][r] = x1;
      rowmax[r] = half16_max(fmaxf(x0, x1));
    }

    float corr[8];
    #pragma unroll
    for (int r = 0; r < 8; ++r) {
      float nm = fmaxf(m[r], rowmax[r]);
      corr[r] = __expf(m[r] - nm);
      m[r] = nm;
    }

    // P = exp(S - m), stash into per-wave LDS in row-major [16 q][32 k]
    #pragma unroll
    for (int r = 0; r < 8; ++r) {
      float x0 = sc[0][r], x1 = sc[1][r];
      float p0 = (x0 <= -1e29f) ? 0.f : __expf(x0 - m[r]);
      float p1 = (x1 <= -1e29f) ? 0.f : __expf(x1 - m[r]);
      int prow = wave * 16 + r + 8 * g;
      Ps[prow * LDP + ln]      = (_Float16)p0;
      Ps[prow * LDP + 16 + ln] = (_Float16)p1;
      l[r] = l[r] * corr[r] + half16_sum(p0 + p1);
    }
    #pragma unroll
    for (int nt = 0; nt < 4; ++nt)
      #pragma unroll
      for (int r = 0; r < 8; ++r) accO[nt][r] *= corr[r];

    Frag16 pf;  // P as A-fragment 16x32
    {
      int row = wave * 16 + ln;
      pf.h[0] = *(const v8h*)&Ps[row * LDP + 8 * g];
      pf.h[1] = *(const v8h*)&Ps[row * LDP + 16 + 8 * g];
    }
    #pragma unroll
    for (int nt = 0; nt < 4; ++nt) {  // O += P(16x32) * V(32x16) per d tile
      Frag16 vf;
      int d = nt * 16 + ln;
      vf.h[0] = *(const v8h*)&Vs[d * LDV + 8 * g];
      vf.h[1] = *(const v8h*)&Vs[d * LDV + 16 + 8 * g];
      accO[nt] = wmma16x16x32(pf.v, vf.v, accO[nt]);
    }
  }

  float* ob = out + (size_t)b * S * 512 + h * 64;
  #pragma unroll
  for (int r = 0; r < 8; ++r) {
    float inv = (l[r] > 0.f) ? (1.f / l[r]) : 0.f;
    int q = q0 + r + 8 * g;
    #pragma unroll
    for (int nt = 0; nt < 4; ++nt)
      ob[(size_t)q * 512 + nt * 16 + ln] = accO[nt][r] * inv;
  }
}

// ---------------------------------------------------------------------------
// Fused residual + LayerNorm: y = LN(a + r) * g + b  -> f32 and f16 mirrors
// one wave per 512-wide row, 8 rows per block
// ---------------------------------------------------------------------------
__global__ __launch_bounds__(256)
void ln_res_kernel(const float* __restrict__ a, const float* __restrict__ r,
                   const float* __restrict__ gw, const float* __restrict__ bw,
                   float* __restrict__ outF, _Float16* __restrict__ outH) {
  const int H = 512;
  int row = blockIdx.x * 8 + (threadIdx.x >> 5);
  int lane = threadIdx.x & 31;
  const float* pa = a + (size_t)row * H;
  const float* pr = r + (size_t)row * H;
  v4f x[4];
  float s = 0.f, sq = 0.f;
  #pragma unroll
  for (int t = 0; t < 4; ++t) {
    int c = t * 128 + lane * 4;
    v4f v = *(const v4f*)(pa + c) + *(const v4f*)(pr + c);
    x[t] = v;
    #pragma unroll
    for (int j = 0; j < 4; ++j) { s += v[j]; sq += v[j] * v[j]; }
  }
  #pragma unroll
  for (int msk = 1; msk < 32; msk <<= 1) {
    s += __shfl_xor(s, msk, 32);
    sq += __shfl_xor(sq, msk, 32);
  }
  float mu = s * (1.f / 512.f);
  float var = sq * (1.f / 512.f) - mu * mu;
  float rstd = rsqrtf(var + 1e-5f);
  #pragma unroll
  for (int t = 0; t < 4; ++t) {
    int c = t * 128 + lane * 4;
    v4f vg = *(const v4f*)(gw + c);
    v4f vb = *(const v4f*)(bw + c);
    v4f y; v4h yh;
    #pragma unroll
    for (int j = 0; j < 4; ++j) {
      y[j] = (x[t][j] - mu) * rstd * vg[j] + vb[j];
      yh[j] = (_Float16)y[j];
    }
    *(v4f*)(outF + (size_t)row * H + c) = y;
    *(v4h*)(outH + (size_t)row * H + c) = yh;
  }
}

// ---------------------------------------------------------------------------
extern "C" void kernel_launch(void* const* d_in, const int* in_sizes, int n_in,
                              void* d_out, int out_size, void* d_ws, size_t ws_size,
                              hipStream_t stream) {
  (void)in_sizes; (void)n_in; (void)out_size; (void)ws_size;
  const int B = 2, S = 2048, H = 512, FF = 2048;
  const int M = B * S;

  const float* x     = (const float*)d_in[0];
  const float* amask = (const float*)d_in[1];
  const float* q1_w = (const float*)d_in[2];  const float* q1_b = (const float*)d_in[3];
  const float* k1_w = (const float*)d_in[4];  const float* k1_b = (const float*)d_in[5];
  const float* v1_w = (const float*)d_in[6];  const float* v1_b = (const float*)d_in[7];
  const float* q2_w = (const float*)d_in[8];  const float* q2_b = (const float*)d_in[9];
  const float* k2_w = (const float*)d_in[10]; const float* k2_b = (const float*)d_in[11];
  const float* v2_w = (const float*)d_in[12]; const float* v2_b = (const float*)d_in[13];
  const float* ln_g = (const float*)d_in[14]; const float* ln_b = (const float*)d_in[15];
  const float* ff1_w = (const float*)d_in[16]; const float* ff1_b = (const float*)d_in[17];
  const float* ff2_w = (const float*)d_in[18]; const float* ff2_b = (const float*)d_in[19];

  char* ws = (char*)d_ws;
  size_t off = 0;
  auto alloc = [&](size_t bytes) -> void* {
    void* p = ws + off;
    off = (off + bytes + 255) & ~(size_t)255;
    return p;
  };
  _Float16* x_h   = (_Float16*)alloc((size_t)M * H * 2);
  _Float16* h1_h  = (_Float16*)alloc((size_t)M * H * 2);
  _Float16* h2_h  = (_Float16*)alloc((size_t)M * H * 2);
  _Float16* qkv_h = (_Float16*)alloc((size_t)M * 3 * H * 2);
  _Float16* w1t   = (_Float16*)alloc((size_t)3 * H * H * 2);   // [1536][512]
  _Float16* w2t   = (_Float16*)alloc((size_t)3 * H * H * 2);
  _Float16* wf1t  = (_Float16*)alloc((size_t)FF * H * 2);      // [2048][512]
  _Float16* wf2t  = (_Float16*)alloc((size_t)H * FF * 2);      // [512][2048]
  float*    bqkv1 = (float*)alloc((size_t)3 * H * 4);
  float*    bqkv2 = (float*)alloc((size_t)3 * H * 4);
  float*    att   = (float*)alloc((size_t)M * H * 4);
  float*    h1f   = (float*)alloc((size_t)M * H * 4);
  float*    h2f   = (float*)alloc((size_t)M * H * 4);
  _Float16* ffh   = (_Float16*)alloc((size_t)M * FF * 2);

  // --- one-time (per call) precision conversion / weight transposes ---
  cvt_f16_kernel<<<(M * H / 4 + 255) / 256, 256, 0, stream>>>(x, x_h, M * H / 4);
  cvt_tr_kernel<<<dim3(H / 16, H / 16), 256, 0, stream>>>(q1_w, w1t, H, H, 0,     H);
  cvt_tr_kernel<<<dim3(H / 16, H / 16), 256, 0, stream>>>(k1_w, w1t, H, H, 512,  H);
  cvt_tr_kernel<<<dim3(H / 16, H / 16), 256, 0, stream>>>(v1_w, w1t, H, H, 1024, H);
  cvt_tr_kernel<<<dim3(H / 16, H / 16), 256, 0, stream>>>(q2_w, w2t, H, H, 0,     H);
  cvt_tr_kernel<<<dim3(H / 16, H / 16), 256, 0, stream>>>(k2_w, w2t, H, H, 512,  H);
  cvt_tr_kernel<<<dim3(H / 16, H / 16), 256, 0, stream>>>(v2_w, w2t, H, H, 1024, H);
  cvt_tr_kernel<<<dim3(H / 16, FF / 16), 256, 0, stream>>>(ff1_w, wf1t, H, FF, 0, H);
  cvt_tr_kernel<<<dim3(FF / 16, H / 16), 256, 0, stream>>>(ff2_w, wf2t, FF, H, 0, FF);
  hipMemcpyAsync(bqkv1,        q1_b, H * 4, hipMemcpyDeviceToDevice, stream);
  hipMemcpyAsync(bqkv1 + 512,  k1_b, H * 4, hipMemcpyDeviceToDevice, stream);
  hipMemcpyAsync(bqkv1 + 1024, v1_b, H * 4, hipMemcpyDeviceToDevice, stream);
  hipMemcpyAsync(bqkv2,        q2_b, H * 4, hipMemcpyDeviceToDevice, stream);
  hipMemcpyAsync(bqkv2 + 512,  k2_b, H * 4, hipMemcpyDeviceToDevice, stream);
  hipMemcpyAsync(bqkv2 + 1024, v2_b, H * 4, hipMemcpyDeviceToDevice, stream);

  // layer 1: fused QKV projection -> flash attn -> residual LN
  gemm_wmma_kernel<<<dim3(3 * H / 128, M / 128), 256, 0, stream>>>(
      x_h, w1t, bqkv1, nullptr, nullptr, qkv_h, M, 3 * H, H, 0);
  flash_attn_kernel<<<dim3(S / 128, 8, B), 256, 0, stream>>>(qkv_h, amask, att, S);
  ln_res_kernel<<<M / 8, 256, 0, stream>>>(att, x, ln_g, ln_b, h1f, h1_h);
  // layer 2
  gemm_wmma_kernel<<<dim3(3 * H / 128, M / 128), 256, 0, stream>>>(
      h1_h, w2t, bqkv2, nullptr, nullptr, qkv_h, M, 3 * H, H, 0);
  flash_attn_kernel<<<dim3(S / 128, 8, B), 256, 0, stream>>>(qkv_h, amask, att, S);
  ln_res_kernel<<<M / 8, 256, 0, stream>>>(att, h1f, ln_g, ln_b, h2f, h2_h);
  // FFN: relu(h2@W1 + b1)@W2 + b2 + h2 -> d_out (residual fused into epilogue)
  gemm_wmma_kernel<<<dim3(FF / 128, M / 128), 256, 0, stream>>>(
      h2_h, wf1t, ff1_b, nullptr, nullptr, ffh, M, FF, H, 1);
  gemm_wmma_kernel<<<dim3(H / 128, M / 128), 256, 0, stream>>>(
      ffh, wf2t, ff2_b, h2f, (float*)d_out, nullptr, M, H, FF, 0);
}
